// GCN_33062658245470
// MI455X (gfx1250) — matrix operand order repaired and checked by reading, not compiled, not run on hardware
//
#include <hip/hip_runtime.h>
#include <cmath>

typedef __attribute__((ext_vector_type(16))) __bf16 v16bf;
typedef __attribute__((ext_vector_type(8)))  __bf16 v8bf;
typedef __attribute__((ext_vector_type(8)))  float  v8f;

#define F_IN 128
#define DIM  64
#define NCLS 10
#define LDS_K 136   // max K (128) + 8 bf16 pad -> 272B row stride, bank-conflict free

// ---------------------------------------------------------------------------
// WMMA bf16 GEMM: Out[M x 64] = A[M x K] @ W[K x 64]   (EPI=1: += bias, ReLU)
// W is transposed+converted to bf16 in LDS once per block; each wave computes a
// 16x64 strip: one A fragment feeds 4 v_wmma_f32_16x16x32_bf16 per K-step, with
// B fragments read as two ds_load_b128 each.
// Requires M % 16 == 0, K % 32 == 0 (true here: M in {100000,512}, K in {128,64}).
// ---------------------------------------------------------------------------
template <int EPI>
__global__ void k_wmma_gemm(const float* __restrict__ A, const float* __restrict__ W,
                            const float* __restrict__ bias, float* __restrict__ Out,
                            int M, int K) {
    __shared__ __bf16 Wt[DIM * LDS_K];
    // cooperative transpose: Wt[col][k] = bf16(W[k][col]); global reads coalesced
    for (int idx = threadIdx.x; idx < K * DIM; idx += blockDim.x) {
        int k = idx >> 6, col = idx & 63;
        Wt[col * LDS_K + k] = (__bf16)W[idx];
    }
    __syncthreads();

    const int wave = threadIdx.x >> 5;
    const int lane = threadIdx.x & 31;
    const int half = lane >> 4;
    const int l16  = lane & 15;
    const int strip = blockIdx.x * (blockDim.x >> 5) + wave;
    if (strip >= (M >> 4)) return;

    const float* arowp = A + (size_t)(strip * 16 + l16) * K;

    v8f acc[4] = {};
    for (int kb = 0; kb < K; kb += 32) {
        // A 16x32 bf16 fragment: lane half selects K-interleave; two contiguous
        // 8-float chunks at kb+8*half and kb+16+8*half (4x b128 global loads)
        const float4* ap0 = (const float4*)(arowp + kb + 8 * half);
        const float4* ap1 = (const float4*)(arowp + kb + 16 + 8 * half);
        float4 c0 = ap0[0], c1 = ap0[1], c2 = ap1[0], c3 = ap1[1];
        v16bf a;
        a[0] = (__bf16)c0.x; a[1] = (__bf16)c0.y; a[2]  = (__bf16)c0.z; a[3]  = (__bf16)c0.w;
        a[4] = (__bf16)c1.x; a[5] = (__bf16)c1.y; a[6]  = (__bf16)c1.z; a[7]  = (__bf16)c1.w;
        a[8] = (__bf16)c2.x; a[9] = (__bf16)c2.y; a[10] = (__bf16)c2.z; a[11] = (__bf16)c2.w;
        a[12] = (__bf16)c3.x; a[13] = (__bf16)c3.y; a[14] = (__bf16)c3.z; a[15] = (__bf16)c3.w;
#pragma unroll
        for (int t = 0; t < 4; ++t) {
            // B 32x16 fragment: lanes 0-15 hold K=kb..kb+7,kb+16..kb+23, lanes 16-31 hold
            // kb+8..kb+15,kb+24..kb+31, column = t*16 + lane%16; 2x b128 from Wt
            const __bf16* bp = Wt + (t * 16 + l16) * LDS_K + kb + 8 * half;
            v8bf blo = *(const v8bf*)bp;
            v8bf bhi = *(const v8bf*)(bp + 16);
            v16bf b = __builtin_shufflevector(blo, bhi,
                0, 1, 2, 3, 4, 5, 6, 7, 8, 9, 10, 11, 12, 13, 14, 15);
            acc[t] = __builtin_amdgcn_wmma_f32_16x16x32_bf16(
                false, a, false, b, (short)0, acc[t], false, false);
        }
    }

    // D layout: VGPR r -> row r + 8*half, col = lane%16
    const int orow0 = strip * 16 + 8 * half;
#pragma unroll
    for (int t = 0; t < 4; ++t) {
        const int ocol = t * 16 + l16;
        const float bv = EPI ? bias[ocol] : 0.f;
#pragma unroll
        for (int r = 0; r < 8; ++r) {
            float v = acc[t][r];
            if (EPI) { v += bv; v = fmaxf(v, 0.f); }
            Out[(size_t)(orow0 + r) * DIM + ocol] = v;
        }
    }
}

// ---------------------------------------------------------------------------
__global__ void k_fill(float* __restrict__ p, float v, int n) {
    int t = blockIdx.x * blockDim.x + threadIdx.x;
    if (t < n) p[t] = v;
}

__global__ void k_degree(const int* __restrict__ dst, const float* __restrict__ ew,
                         float* __restrict__ deg, int E) {
    int t = blockIdx.x * blockDim.x + threadIdx.x;
    if (t >= E) return;
    int d = __builtin_nontemporal_load(&dst[t]);      // edge data streamed once:
    float w = __builtin_nontemporal_load(&ew[t]);     // keep L2 for node features
    atomicAdd(&deg[d], w);
}

__global__ void k_rsqrt_inplace(float* __restrict__ d, int n) {
    int t = blockIdx.x * blockDim.x + threadIdx.x;
    if (t < n) d[t] = rsqrtf(d[t]);
}

// AGG[n] = dinv[n]^2 * H[n]   (self-loop term; also initializes AGG)
__global__ void k_self_init(const float* __restrict__ H, const float* __restrict__ dinv,
                            float* __restrict__ AGG, int N) {
    int t = blockIdx.x * blockDim.x + threadIdx.x;
    if (t >= N * 16) return;
    int n = t >> 4, q = t & 15;
    float di = dinv[n];
    float s = di * di;
    float4 v = ((const float4*)(H + (size_t)n * DIM))[q];
    v.x *= s; v.y *= s; v.z *= s; v.w *= s;
    ((float4*)(AGG + (size_t)n * DIM))[q] = v;
}

// AGG[dst] += dinv[src]*ew*dinv[dst] * H[src]; 4 threads per edge (16 feats each)
__global__ void k_scatter(const int* __restrict__ src, const int* __restrict__ dst,
                          const float* __restrict__ ew, const float* __restrict__ dinv,
                          const float* __restrict__ H, float* __restrict__ AGG, int E) {
    unsigned t = blockIdx.x * blockDim.x + threadIdx.x;
    if (t >= (unsigned)E * 4u) return;
    int e = (int)(t >> 2), q = (int)(t & 3);
    int s = __builtin_nontemporal_load(&src[e]);
    int d = __builtin_nontemporal_load(&dst[e]);
    float w = __builtin_nontemporal_load(&ew[e]);
    float nrm = dinv[s] * w * dinv[d];
    const float4* hp = (const float4*)(H + (size_t)s * DIM) + q * 4;
    float* op = AGG + (size_t)d * DIM + q * 16;
#pragma unroll
    for (int j = 0; j < 4; ++j) {
        float4 v = hp[j];
        atomicAdd(op + 4 * j + 0, nrm * v.x);
        atomicAdd(op + 4 * j + 1, nrm * v.y);
        atomicAdd(op + 4 * j + 2, nrm * v.z);
        atomicAdd(op + 4 * j + 3, nrm * v.w);
    }
}

// hout[n] = relu( (AGG[n] + b) / max(||AGG[n]+b||_2, 1e-12) )
__global__ void k_finalize(const float* __restrict__ AGG, const float* __restrict__ b,
                           float* __restrict__ hout, int N) {
    int n = blockIdx.x * blockDim.x + threadIdx.x;
    if (n >= N) return;
    const float4* ap = (const float4*)(AGG + (size_t)n * DIM);
    const float4* bp = (const float4*)b;
    float4 v[16];
    float ss = 0.f;
#pragma unroll
    for (int q = 0; q < 16; ++q) {
        float4 x = ap[q], bb = bp[q];
        x.x += bb.x; x.y += bb.y; x.z += bb.z; x.w += bb.w;
        ss += x.x * x.x + x.y * x.y + x.z * x.z + x.w * x.w;
        v[q] = x;
    }
    float inv = 1.f / fmaxf(sqrtf(ss), 1e-12f);
    float4* op = (float4*)(hout + (size_t)n * DIM);
#pragma unroll
    for (int q = 0; q < 16; ++q) {
        float4 x = v[q];
        x.x = fmaxf(x.x * inv, 0.f);
        x.y = fmaxf(x.y * inv, 0.f);
        x.z = fmaxf(x.z * inv, 0.f);
        x.w = fmaxf(x.w * inv, 0.f);
        op[q] = x;
    }
}

// pooled[batch[n]] += h[n]; 4 threads per node (16 feats each)
__global__ void k_pool(const float* __restrict__ h, const int* __restrict__ batch,
                       float* __restrict__ pooled, int N) {
    int t = blockIdx.x * blockDim.x + threadIdx.x;
    if (t >= N * 4) return;
    int n = t >> 2, q = t & 3;
    int g = batch[n];
    const float4* hp = (const float4*)(h + (size_t)n * DIM) + q * 4;
    float* op = pooled + (size_t)g * DIM + q * 16;
#pragma unroll
    for (int j = 0; j < 4; ++j) {
        float4 v = hp[j];
        atomicAdd(op + 4 * j + 0, v.x);
        atomicAdd(op + 4 * j + 1, v.y);
        atomicAdd(op + 4 * j + 2, v.z);
        atomicAdd(op + 4 * j + 3, v.w);
    }
}

// out[g] = log_softmax( z[g] @ fc2_w + fc2_b )   (64 -> 10 per graph)
__global__ void k_head(const float* __restrict__ z, const float* __restrict__ W,
                       const float* __restrict__ b, float* __restrict__ out, int G) {
    int g = blockIdx.x * blockDim.x + threadIdx.x;
    if (g >= G) return;
    float lg[NCLS];
#pragma unroll
    for (int c = 0; c < NCLS; ++c) lg[c] = b[c];
    const float* zr = z + (size_t)g * DIM;
    for (int k = 0; k < DIM; ++k) {
        float zv = zr[k];
        const float* wr = W + (size_t)k * NCLS;
#pragma unroll
        for (int c = 0; c < NCLS; ++c) lg[c] += zv * wr[c];
    }
    float mx = lg[0];
#pragma unroll
    for (int c = 1; c < NCLS; ++c) mx = fmaxf(mx, lg[c]);
    float se = 0.f;
#pragma unroll
    for (int c = 0; c < NCLS; ++c) se += expf(lg[c] - mx);
    float lse = logf(se) + mx;
#pragma unroll
    for (int c = 0; c < NCLS; ++c) out[(size_t)g * NCLS + c] = lg[c] - lse;
}

// ---------------------------------------------------------------------------
extern "C" void kernel_launch(void* const* d_in, const int* in_sizes, int n_in,
                              void* d_out, int out_size, void* d_ws, size_t ws_size,
                              hipStream_t stream) {
    const float* x     = (const float*)d_in[0];
    const int*   ei    = (const int*)  d_in[1];
    const int*   batch = (const int*)  d_in[2];
    const float* ew    = (const float*)d_in[3];
    const float* W1 = (const float*)d_in[4],  *b1 = (const float*)d_in[5];
    const float* W2 = (const float*)d_in[6],  *b2 = (const float*)d_in[7];
    const float* W3 = (const float*)d_in[8],  *b3 = (const float*)d_in[9];
    const float* fc1w = (const float*)d_in[10], *fc1b = (const float*)d_in[11];
    const float* fc2w = (const float*)d_in[12], *fc2b = (const float*)d_in[13];
    float* out = (float*)d_out;

    const int N = in_sizes[0] / F_IN;
    const int E = in_sizes[1] / 2;
    const int G = out_size / NCLS;
    const int* src = ei;
    const int* dst = ei + E;

    // workspace carve-up (all float, base 256B-aligned by hipMalloc)
    float* ws = (float*)d_ws;
    float* dinv = ws;  ws += (size_t)((N + 3) & ~3);   // degrees -> rsqrt in place
    float* H    = ws;  ws += (size_t)N * DIM;          // X@W
    float* AGG  = ws;  ws += (size_t)N * DIM;          // scatter accumulator
    float* hcur = ws;  ws += (size_t)N * DIM;          // layer activations
    float* pooled = ws; ws += (size_t)G * DIM;
    float* zbuf = ws;  ws += (size_t)G * DIM;

    const int TB = 256;
    // --- symmetric normalization: deg = 1 (self loop) + sum(ew at dst); dinv = rsqrt
    k_fill<<<(N + TB - 1) / TB, TB, 0, stream>>>(dinv, 1.0f, N);
    k_degree<<<(E + TB - 1) / TB, TB, 0, stream>>>(dst, ew, dinv, E);
    k_rsqrt_inplace<<<(N + TB - 1) / TB, TB, 0, stream>>>(dinv, N);

    auto layer = [&](const float* hin, int K, const float* W, const float* b) {
        int strips = N / 16;                       // 8 waves/block, 16x64 strip/wave
        k_wmma_gemm<0><<<(strips + 7) / 8, 256, 0, stream>>>(hin, W, nullptr, H, N, K);
        k_self_init<<<(N * 16 + TB - 1) / TB, TB, 0, stream>>>(H, dinv, AGG, N);
        unsigned sc = (unsigned)E * 4u;
        k_scatter<<<(sc + TB - 1) / TB, TB, 0, stream>>>(src, dst, ew, dinv, H, AGG, E);
        k_finalize<<<(N + TB - 1) / TB, TB, 0, stream>>>(AGG, b, hcur, N);
    };
    layer(x,    F_IN, W1, b1);   // hcur written only after the gemm consumed hin,
    layer(hcur, DIM,  W2, b2);   // so reusing hcur in/out across kernels is safe
    layer(hcur, DIM,  W3, b3);

    // --- global_add_pool
    k_fill<<<(G * DIM + TB - 1) / TB, TB, 0, stream>>>(pooled, 0.0f, G * DIM);
    k_pool<<<(N * 4 + TB - 1) / TB, TB, 0, stream>>>(hcur, batch, pooled, N);

    // --- fc1 (+bias, ReLU) via WMMA, then fc2 + log_softmax
    int strips2 = G / 16;
    k_wmma_gemm<1><<<(strips2 + 7) / 8, 256, 0, stream>>>(pooled, fc1w, fc1b, zbuf, G, DIM);
    k_head<<<(G + 63) / 64, 64, 0, stream>>>(zbuf, fc2w, fc2b, out, G);
}
